// LASLNNet_29970281792120
// MI455X (gfx1250) — matrix operand order repaired and checked by reading, not compiled, not run on hardware
//
#include <hip/hip_runtime.h>

typedef __attribute__((ext_vector_type(16))) _Float16 v16h;
typedef __attribute__((ext_vector_type(8)))  _Float16 v8h;
typedef __attribute__((ext_vector_type(2)))  _Float16 v2h;
typedef __attribute__((ext_vector_type(8)))  float    v8f;

#define WAVE 32
#define FC_KC 65536L

constexpr int ctz_c(int x) { return (x & 1) ? 0 : 1 + ctz_c(x >> 1); }

// ---- f32 input (B,2,1,spatial) -> f16 activations laid out (b, spatial, c) , c = ri (Cin=1) ----
__global__ void cvt_input_kernel(const float* __restrict__ x,
                                 _Float16* __restrict__ act,
                                 long S, long total) {
  long idx = (long)blockIdx.x * blockDim.x + threadIdx.x;
  if (idx >= total) return;
  const int ri = (int)(idx & 1);
  const long q = idx >> 1;          // q = b*S + s
  const long s = q % S;
  const int  b = (int)(q / S);
  act[idx] = (_Float16)x[((long)b * 2 + ri) * S + s];
}

// ---- pack complex weights into f16 GEMM matrix W[n][k], K-contiguous per column ----
// K = tap16*(2*Cin) + (ri*Cin + cin), tap16 = t1*8+t2*4+t3*2+t4
// columns: [0,Cout)=real out, [Cout,2Cout)=imag out  (zero-padded to Npad)
__global__ void pack_weights_kernel(const float* __restrict__ wr,
                                    const float* __restrict__ wi,
                                    _Float16* __restrict__ Wp,
                                    int Cin, int Cout, int Npad) {
  const int Kl = 32 * Cin;
  const int c2 = 2 * Cin;
  int idx = blockIdx.x * blockDim.x + threadIdx.x;
  if (idx >= Npad * Kl) return;
  const int nn = idx / Kl;
  const int k  = idx % Kl;
  float v = 0.f;
  if (nn < 2 * Cout) {
    const int tap = k / c2;
    const int c   = k % c2;
    const int riIn = c / Cin;
    const int cin  = c % Cin;
    const int t1 = (tap >> 3) & 1, t2 = (tap >> 2) & 1;
    const int t3 = (tap >> 1) & 1, t4 = tap & 1;
    const int ro = (nn >= Cout) ? 1 : 0;
    const int co = ro ? (nn - Cout) : nn;
    const int widx = ((((co * Cin + cin) * 2 + t1) * 2 + t2) * 2 + t3) * 2 + t4;
    const float a = wr[widx], cc = wi[widx];
    // yr = xr*wr - xi*wi ; yi = xr*wi + xi*wr
    v = ro ? (riIn ? a : cc) : (riIn ? -cc : a);
  }
  Wp[idx] = (_Float16)v;
}

// ---- implicit-GEMM complex conv: one wave computes a 16(M) x 16(N) tile ----
// Activation layout (b,d1..d4,c) + K = tap*(2CIN)+c makes A-fragments contiguous:
// CIN>=4 -> two aligned b128 loads per lane per WMMA; CIN<4 -> aligned u32 pairs.
template <int CIN>
__global__ __launch_bounds__(WAVE) void conv_wmma_kernel(
    const _Float16* __restrict__ actIn,
    const _Float16* __restrict__ Wp,
    const float* __restrict__ br, const float* __restrict__ bi,
    _Float16* __restrict__ actOut,
    int B, int Cout,
    int I1, int I2, int I3, int I4,
    int O1, int O2, int O3, int O4,
    int Mtotal) {
  constexpr int C2   = 2 * CIN;
  constexpr int C2SH = ctz_c(C2);
  constexpr int Kl   = 32 * CIN;

  const int lane = threadIdx.x;
  const int ncol = blockIdx.y * 16 + (lane & 15);

  // A-row this lane gathers (clamped; invalid rows masked at store)
  int m = blockIdx.x * 16 + (lane & 15);
  if (m >= Mtotal) m = Mtotal - 1;
  int t = m;
  const int o4 = t % O4; t /= O4;
  const int o3 = t % O3; t /= O3;
  const int o2 = t % O2; t /= O2;
  const int o1 = t % O1; t /= O1;
  const int b  = t;

  // element strides of (b, d1, d2, d3, d4, c) layout; all multiples of C2
  const long p4 = C2;
  const long p3 = (long)I4 * p4;
  const long p2 = (long)I3 * p3;
  const long p1 = (long)I2 * p2;
  const long pB = (long)I1 * p1;
  const long base = (long)b * pB + (long)o1 * p1 + (long)o2 * p2 +
                    (long)o3 * p3 + (long)o4 * p4;

  auto toff = [&](int tp) -> long {   // tap16 -> position offset
    return (long)((tp >> 3) & 1) * p1 + (long)((tp >> 2) & 1) * p2 +
           (long)((tp >> 1) & 1) * p3 + (long)(tp & 1) * p4;
  };

  // ISA f16 A 16x32 layout: lanes 0-15 hold K 0-7 & 16-23; lanes 16-31 hold 8-15 & 24-31
  const int Kb = (lane < 16) ? 0 : 8;
  const _Float16* wcol = Wp + (long)ncol * Kl + Kb;

  v8f acc = {0.f, 0.f, 0.f, 0.f, 0.f, 0.f, 0.f, 0.f};

#pragma unroll
  for (int ks = 0; ks < CIN; ++ks) {
    v16h a;
    if constexpr (CIN >= 4) {
      const int kA = ks * 32 + Kb;        // K of a[0]
      const int kB = kA + 16;             // K of a[8]
      const int tapA = kA >> C2SH, cA = kA & (C2 - 1);
      const int tapB = kB >> C2SH, cB = kB & (C2 - 1);
      const v8h lo = *(const v8h*)(actIn + base + toff(tapA) + cA);  // 16B aligned
      const v8h hi = *(const v8h*)(actIn + base + toff(tapB) + cB);
#pragma unroll
      for (int j = 0; j < 8; ++j) { a[j] = lo[j]; a[8 + j] = hi[j]; }
    } else {
#pragma unroll
      for (int i = 0; i < 8; ++i) {       // aligned u32 K-pairs
        const int kk = ks * 32 + ((i < 4) ? (Kb + 2 * i) : (16 + Kb + 2 * (i - 4)));
        const int tp = kk >> C2SH, cc = kk & (C2 - 1);
        const v2h pr = *(const v2h*)(actIn + base + toff(tp) + cc);
        a[2 * i]     = pr[0];
        a[2 * i + 1] = pr[1];
      }
    }
    const v8h blo = *(const v8h*)(wcol + ks * 32);        // 16B aligned
    const v8h bhi = *(const v8h*)(wcol + ks * 32 + 16);
    v16h bf;
#pragma unroll
    for (int j = 0; j < 8; ++j) { bf[j] = blo[j]; bf[8 + j] = bhi[j]; }
    __builtin_prefetch((const void*)(wcol + (ks + 1) * 32), 0, 0);
    acc = __builtin_amdgcn_wmma_f32_16x16x32_f16(
        false, a, false, bf, (short)0, acc, false, false);
  }

  // epilogue: bias + ReLU + f16 store; output layout (pos, c) -> off = mr*2Cout + ncol
  const bool colValid = ncol < 2 * Cout;
  const int ro = (ncol >= Cout) ? 1 : 0;
  const int co = ro ? (ncol - Cout) : ncol;
  const float bias = colValid ? (ro ? bi[co] : br[co]) : 0.f;
  const int rowAdd = (lane < 16) ? 0 : 8;   // C/D map: M = r + 8*(lane>=16), N = lane&15

#pragma unroll
  for (int r = 0; r < 8; ++r) {
    const int mr = blockIdx.x * 16 + r + rowAdd;
    if (colValid && mr < Mtotal) {
      float v = acc[r] + bias;
      v = v > 0.f ? v : 0.f;
      actOut[(long)mr * (2 * Cout) + ncol] = (_Float16)v;
    }
  }
}

// ---- FC stage 1: deterministic partial dot-products (no atomics, graph-safe) ----
// h lives in (b, spatial, c) layout with c = ri*64+co; reference k = co*OS + sp.
__global__ __launch_bounds__(256) void fc_partial_kernel(
    const _Float16* __restrict__ h, const float* __restrict__ fcw,
    float* __restrict__ partial, long N, long OS, int B) {
  const int chunk = blockIdx.x;
  const int row = blockIdx.y;                 // row = ri*B + b (16 rows)
  const int ri = row / B;
  const int b  = row % B;
  const long k0 = (long)chunk * FC_KC;
  long kend = k0 + FC_KC; if (kend > N) kend = N;
  float acc[8] = {0.f,0.f,0.f,0.f,0.f,0.f,0.f,0.f};
  for (long k = k0 + threadIdx.x; k < kend; k += 256) {
    const long co = k / OS;
    const long sp = k % OS;
    const float hv = (float)h[((long)b * OS + sp) * 128 + (long)ri * 64 + co];
#pragma unroll
    for (int j = 0; j < 8; ++j) acc[j] += hv * fcw[(long)j * N + k];
  }
  __shared__ float red[256 * 8];
#pragma unroll
  for (int j = 0; j < 8; ++j) red[threadIdx.x * 8 + j] = acc[j];
  __syncthreads();
  for (int s = 128; s > 0; s >>= 1) {
    if ((int)threadIdx.x < s) {
#pragma unroll
      for (int j = 0; j < 8; ++j)
        red[threadIdx.x * 8 + j] += red[(threadIdx.x + s) * 8 + j];
    }
    __syncthreads();
  }
  if (threadIdx.x == 0) {
#pragma unroll
    for (int j = 0; j < 8; ++j)
      partial[((long)chunk * 16 + row) * 8 + j] = red[j];
  }
}

// ---- FC stage 2: reduce partials + bias ----
__global__ void fc_final_kernel(const float* __restrict__ partial,
                                const float* __restrict__ fcb,
                                float* __restrict__ out, int nchunks, int B) {
  const int i = threadIdx.x;
  if (i >= 128) return;
  const int row = i >> 3;
  const int j = i & 7;
  float s = fcb[j];
  for (int c = 0; c < nchunks; ++c) s += partial[((long)c * 16 + row) * 8 + j];
  const int ri = row / B;
  const int b  = row % B;
  out[(b * 2 + ri) * 8 + j] = s;   // output (B, 2, 8)
}

extern "C" void kernel_launch(void* const* d_in, const int* in_sizes, int n_in,
                              void* d_out, int out_size, void* d_ws, size_t ws_size,
                              hipStream_t stream) {
  const int B = 8;
  static const int chans[7] = {1, 2, 4, 8, 16, 32, 64};
  int I1 = 64, I2 = 64, I3 = 11, I4 = 11;

  // workspace layout (all 256B-aligned)
  char* ws = (char*)d_ws;
  const size_t ACTA_BYTES = 172236800;  // largest even-stage act (act6, f16)
  const size_t ACTB_BYTES = 128323584;  // largest odd-stage act (act5, f16)
  _Float16* actA  = (_Float16*)ws;
  _Float16* actB  = (_Float16*)(ws + ACTA_BYTES);
  _Float16* WpAll = (_Float16*)(ws + ACTA_BYTES + ACTB_BYTES);
  float*    partial = (float*)(ws + ACTA_BYTES + ACTB_BYTES + 512 * 1024);

  // 1) input conversion: f32 (B,2,1,spatial) -> f16 (b, spatial, ri)
  {
    const float* x = (const float*)d_in[0];
    long S = (long)I1 * I2 * I3 * I4;
    long total = 2L * B * S;
    int blocks = (int)((total + 255) / 256);
    cvt_input_kernel<<<blocks, 256, 0, stream>>>(x, actA, S, total);
  }

  // 2) six complex conv layers as WMMA implicit GEMMs
  long wpOff = 0;
  _Float16* src = actA;
  _Float16* dst = actB;
  for (int l = 0; l < 6; ++l) {
    const int Cin = chans[l], Cout = chans[l + 1];
    const int Nl = 2 * Cout;
    const int Npad = (Nl < 16) ? 16 : Nl;
    const int Kl = 32 * Cin;
    const float* wr = (const float*)d_in[1 + 4 * l];
    const float* wi = (const float*)d_in[2 + 4 * l];
    const float* br = (const float*)d_in[3 + 4 * l];
    const float* bi = (const float*)d_in[4 + 4 * l];
    _Float16* Wp = WpAll + wpOff;
    wpOff += (long)Npad * Kl;

    {
      int total = Npad * Kl;
      pack_weights_kernel<<<(total + 255) / 256, 256, 0, stream>>>(
          wr, wi, Wp, Cin, Cout, Npad);
    }

    const int O1 = I1 - 1, O2 = I2 - 1, O3 = I3 - 1, O4 = I4 - 1;
    const int Mtotal = B * O1 * O2 * O3 * O4;
    const int Mtiles = (Mtotal + 15) / 16;
    const int Ntiles = Npad / 16;
    dim3 grid(Mtiles, Ntiles);

#define CONV_ARGS src, Wp, br, bi, dst, B, Cout, I1, I2, I3, I4, O1, O2, O3, O4, Mtotal
    switch (Cin) {
      case 1:  conv_wmma_kernel<1><<<grid, WAVE, 0, stream>>>(CONV_ARGS);  break;
      case 2:  conv_wmma_kernel<2><<<grid, WAVE, 0, stream>>>(CONV_ARGS);  break;
      case 4:  conv_wmma_kernel<4><<<grid, WAVE, 0, stream>>>(CONV_ARGS);  break;
      case 8:  conv_wmma_kernel<8><<<grid, WAVE, 0, stream>>>(CONV_ARGS);  break;
      case 16: conv_wmma_kernel<16><<<grid, WAVE, 0, stream>>>(CONV_ARGS); break;
      case 32: conv_wmma_kernel<32><<<grid, WAVE, 0, stream>>>(CONV_ARGS); break;
    }
#undef CONV_ARGS

    _Float16* tmp = src; src = dst; dst = tmp;
    I1 = O1; I2 = O2; I3 = O3; I4 = O4;
  }

  // 3) FC: (16 rows) x (N) @ (N x 8) + bias, two-stage deterministic reduction
  const float* fcw = (const float*)d_in[25];
  const float* fcb = (const float*)d_in[26];
  const long OS = (long)I1 * I2 * I3 * I4;            // 58*58*5*5
  const long N = (long)chans[6] * OS;                 // 64*OS
  const int nchunks = (int)((N + FC_KC - 1) / FC_KC);
  dim3 g2(nchunks, 16);
  fc_partial_kernel<<<g2, 256, 0, stream>>>(src, fcw, partial, N, OS, B);
  fc_final_kernel<<<1, 128, 0, stream>>>(partial, fcb, (float*)d_out, nchunks, B);
}
